// HybridModel_36086315221038
// MI455X (gfx1250) — compile-verified, matrix operand order checked by reading, not run on hardware
//
#include <hip/hip_runtime.h>
#include <hip/hip_bf16.h>

// ---------------------------------------------------------------------------
// Types for CDNA5 WMMA (wave32): bf16 A/B fragments = 16 x bf16 (8 VGPRs),
// f32 C/D accumulator = 8 x f32 (8 VGPRs).
// ---------------------------------------------------------------------------
typedef __attribute__((ext_vector_type(16))) __bf16 v16bf;
typedef __attribute__((ext_vector_type(8)))  float  v8f;

#define B_ROWS 2048
#define EPSV   1e-5f

// fp32 -> bf16 round-to-nearest-even
__device__ inline unsigned short f2bf(float f) {
    union { float f; unsigned int u; } v; v.f = f;
    unsigned int u = v.u;
    unsigned int r = u + 0x7FFFu + ((u >> 16) & 1u);
    return (unsigned short)(r >> 16);
}

// ---------------------------------------------------------------------------
// Pack fp32 [M,K] row-major activations into the WMMA A-fragment layout:
// dst[((mt*Ksteps + ks)*32 + lane)*16 + e] where
//   m = mt*16 + (lane&15)
//   k = ks*32 + (e>>3)*16 + (lane>>4)*8 + (e&7)
// ---------------------------------------------------------------------------
__global__ void pack_a_kernel(const float* __restrict__ src,
                              unsigned short* __restrict__ dst,
                              int K, int Ksteps, int total) {
    int tid = blockIdx.x * blockDim.x + threadIdx.x;
    if (tid >= total) return;
    int e    = tid & 15;
    int lane = (tid >> 4) & 31;
    int t2   = tid >> 9;
    int ks   = t2 % Ksteps;
    int mt   = t2 / Ksteps;
    int m = mt * 16 + (lane & 15);
    int k = ks * 32 + ((e >> 3) << 4) + ((lane >> 4) << 3) + (e & 7);
    dst[tid] = f2bf(src[m * K + k]);
}

// Same as pack_a but applies batch-norm: g*(h-mean)*rstd + beta, then bf16.
__global__ void bn_pack_kernel(const float* __restrict__ src,
                               const float* __restrict__ g,
                               const float* __restrict__ be,
                               const float* __restrict__ mean,
                               const float* __restrict__ rstd,
                               unsigned short* __restrict__ dst,
                               int K, int Ksteps, int total) {
    int tid = blockIdx.x * blockDim.x + threadIdx.x;
    if (tid >= total) return;
    int e    = tid & 15;
    int lane = (tid >> 4) & 31;
    int t2   = tid >> 9;
    int ks   = t2 % Ksteps;
    int mt   = t2 / Ksteps;
    int m = mt * 16 + (lane & 15);
    int k = ks * 32 + ((e >> 3) << 4) + ((lane >> 4) << 3) + (e & 7);
    float v = g[k] * (src[m * K + k] - mean[k]) * rstd[k] + be[k];
    dst[tid] = f2bf(v);
}

// ---------------------------------------------------------------------------
// Pack fp32 [K, Nreal] row-major weights into the WMMA B-fragment layout
// (mirror of A with N in place of M). Columns >= Nreal are zero-padded.
// ---------------------------------------------------------------------------
__global__ void pack_b_kernel(const float* __restrict__ src,
                              unsigned short* __restrict__ dst,
                              int Nreal, int Ksteps, int total) {
    int tid = blockIdx.x * blockDim.x + threadIdx.x;
    if (tid >= total) return;
    int e    = tid & 15;
    int lane = (tid >> 4) & 31;
    int t2   = tid >> 9;
    int ks   = t2 % Ksteps;
    int nt   = t2 / Ksteps;
    int n = nt * 16 + (lane & 15);
    int k = ks * 32 + ((e >> 3) << 4) + ((lane >> 4) << 3) + (e & 7);
    float v = (n < Nreal) ? src[k * Nreal + n] : 0.0f;
    dst[tid] = f2bf(v);
}

// ---------------------------------------------------------------------------
// WMMA GEMM: C[M, Nstride] = A @ B + bias (optional ReLU).
// One 16x16 output tile per wave; 8 waves per block. Operands are already
// in fragment layout -> inner loop = 2 packed loads + v_wmma_f32_16x16x32_bf16.
// ---------------------------------------------------------------------------
__global__ void __launch_bounds__(256)
gemm_wmma_kernel(const unsigned short* __restrict__ A,
                 const unsigned short* __restrict__ Bm,
                 const float* __restrict__ bias,
                 float* __restrict__ C,
                 int Ntiles, int Ksteps, int Nstride, int Nbias, int relu) {
    int wave = threadIdx.x >> 5;
    int lane = threadIdx.x & 31;
    int tile = blockIdx.x * 8 + wave;
    int mt = tile / Ntiles;
    int nt = tile % Ntiles;

    const v16bf* Ap = reinterpret_cast<const v16bf*>(A) + (size_t)(mt * Ksteps) * 32 + lane;
    const v16bf* Bp = reinterpret_cast<const v16bf*>(Bm) + (size_t)(nt * Ksteps) * 32 + lane;

    v8f acc = {};
    for (int ks = 0; ks < Ksteps; ++ks) {
        v16bf a = Ap[ks * 32];
        v16bf b = Bp[ks * 32];
        acc = __builtin_amdgcn_wmma_f32_16x16x32_bf16(
            false, a, false, b, (short)0, acc, false, false);
    }

    int m = mt * 16 + ((lane >> 4) << 3);   // VGPR r holds row m+r
    int n = nt * 16 + (lane & 15);
    float bv = (n < Nbias) ? bias[n] : 0.0f;
#pragma unroll
    for (int r = 0; r < 8; ++r) {
        float v = acc[r] + bv;
        if (relu) v = fmaxf(v, 0.0f);
        C[(size_t)(m + r) * Nstride + n] = v;
    }
}

// ---------------------------------------------------------------------------
// Per-column mean / rstd over the batch (one block per feature).
// ---------------------------------------------------------------------------
__global__ void __launch_bounds__(256)
col_stats_kernel(const float* __restrict__ h,
                 float* __restrict__ mean, float* __restrict__ rstd, int N) {
    int f = blockIdx.x;
    float s = 0.0f, sq = 0.0f;
    for (int r = threadIdx.x; r < B_ROWS; r += 256) {
        float v = h[(size_t)r * N + f];
        s += v; sq += v * v;
    }
    __shared__ float ls[256], lq[256];
    ls[threadIdx.x] = s; lq[threadIdx.x] = sq;
    __syncthreads();
    for (int o = 128; o > 0; o >>= 1) {
        if (threadIdx.x < o) {
            ls[threadIdx.x] += ls[threadIdx.x + o];
            lq[threadIdx.x] += lq[threadIdx.x + o];
        }
        __syncthreads();
    }
    if (threadIdx.x == 0) {
        float m = ls[0] * (1.0f / B_ROWS);
        float v = lq[0] * (1.0f / B_ROWS) - m * m;
        mean[f] = m;
        rstd[f] = rsqrtf(v + EPSV);
    }
}

// ---------------------------------------------------------------------------
// 12-qubit statevector simulation entirely in LDS (32 KB state per block).
// One block (8 waves) per batch element. Qubit q lives at bit (11-q).
// Fuses the final  zexp @ Wp + bp  projection into the probability reduction.
// ---------------------------------------------------------------------------
__global__ void __launch_bounds__(256)
quantum_kernel(const float* __restrict__ params,   // [B, 112] (108 valid)
               const float* __restrict__ Wp,       // [12]
               const float* __restrict__ bp,       // [1]
               float* __restrict__ out) {          // [B]
    __shared__ float re[4096];
    __shared__ float im[4096];
    __shared__ float gates[36 * 8];
    __shared__ float red[256];

    const int b = blockIdx.x, tid = threadIdx.x;

    for (int i = tid; i < 4096; i += 256) { re[i] = 0.0f; im[i] = 0.0f; }
    if (tid == 0) re[0] = 1.0f;

    // 36 threads each build one U = RZ(t2) @ RY(t1) @ RX(t0)
    if (tid < 36) {
        int d = tid / 12, q = tid % 12;
        const float* a = params + (size_t)b * 112 + (d * 12 + q) * 3;
        float t0 = a[0], t1 = a[1], t2 = a[2];
        float c0 = cosf(0.5f * t0), s0 = sinf(0.5f * t0);
        float c1 = cosf(0.5f * t1), s1 = sinf(0.5f * t1);
        float er = cosf(0.5f * t2), ei = -sinf(0.5f * t2);   // e = exp(-i t2/2)
        // m = RY @ RX
        float m00r =  c1 * c0, m00i =  s1 * s0;
        float m01r = -s1 * c0, m01i = -c1 * s0;
        float m10r =  s1 * c0, m10i = -c1 * s0;
        float m11r =  c1 * c0, m11i = -s1 * s0;
        float* g = &gates[tid * 8];
        g[0] = er * m00r - ei * m00i;  g[1] = er * m00i + ei * m00r;
        g[2] = er * m01r - ei * m01i;  g[3] = er * m01i + ei * m01r;
        g[4] = er * m10r + ei * m10i;  g[5] = er * m10i - ei * m10r;  // conj(e)
        g[6] = er * m11r + ei * m11i;  g[7] = er * m11i - ei * m11r;
    }
    __syncthreads();

    for (int d = 0; d < 3; ++d) {
        // single-qubit rotations
        for (int q = 0; q < 12; ++q) {
            const float* U = &gates[(d * 12 + q) * 8];
            float u00r = U[0], u00i = U[1], u01r = U[2], u01i = U[3];
            float u10r = U[4], u10i = U[5], u11r = U[6], u11i = U[7];
            int sh = 11 - q;
            int st = 1 << sh;
            for (int p = tid; p < 2048; p += 256) {
                int i0 = ((p >> sh) << (sh + 1)) | (p & (st - 1));
                int i1 = i0 | st;
                float a0r = re[i0], a0i = im[i0];
                float a1r = re[i1], a1i = im[i1];
                re[i0] = u00r * a0r - u00i * a0i + u01r * a1r - u01i * a1i;
                im[i0] = u00r * a0i + u00i * a0r + u01r * a1i + u01i * a1r;
                re[i1] = u10r * a0r - u10i * a0i + u11r * a1r - u11i * a1i;
                im[i1] = u10r * a0i + u10i * a0r + u11r * a1i + u11i * a1r;
            }
            __syncthreads();
        }
        // CNOT ring: control q -> target (q+1)%12
        for (int q = 0; q < 12; ++q) {
            int cm = 1 << (11 - q);
            int tm = 1 << (11 - ((q + 1) % 12));
            for (int i = tid; i < 4096; i += 256) {
                if ((i & cm) && !(i & tm)) {
                    int j = i | tm;
                    float tr = re[i], ti = im[i];
                    re[i] = re[j]; im[i] = im[j];
                    re[j] = tr;    im[j] = ti;
                }
            }
            __syncthreads();
        }
    }

    // out[b] = sum_i |psi_i|^2 * sum_q sign_q(i) * Wp[q]  + bp
    float wq[12];
#pragma unroll
    for (int q = 0; q < 12; ++q) wq[q] = Wp[q];
    float partial = 0.0f;
    for (int i = tid; i < 4096; i += 256) {
        float pr = re[i] * re[i] + im[i] * im[i];
        float w = 0.0f;
#pragma unroll
        for (int q = 0; q < 12; ++q)
            w += ((i >> (11 - q)) & 1) ? -wq[q] : wq[q];
        partial += pr * w;
    }
    red[tid] = partial;
    __syncthreads();
    for (int o = 128; o > 0; o >>= 1) {
        if (tid < o) red[tid] += red[tid + o];
        __syncthreads();
    }
    if (tid == 0) out[b] = red[0] + bp[0];
}

// ---------------------------------------------------------------------------
// Host orchestration
// ---------------------------------------------------------------------------
static inline char* ws_take(char*& p, size_t bytes) {
    char* r = p;
    p += (bytes + 255) & ~(size_t)255;
    return r;
}

extern "C" void kernel_launch(void* const* d_in, const int* in_sizes, int n_in,
                              void* d_out, int out_size, void* d_ws, size_t ws_size,
                              hipStream_t stream) {
    (void)in_sizes; (void)n_in; (void)out_size; (void)ws_size;
    const float* x  = (const float*)d_in[0];
    const float* W1 = (const float*)d_in[1];
    const float* b1 = (const float*)d_in[2];
    const float* g1 = (const float*)d_in[3];
    const float* be1= (const float*)d_in[4];
    const float* W2 = (const float*)d_in[5];
    const float* b2 = (const float*)d_in[6];
    const float* g2 = (const float*)d_in[7];
    const float* be2= (const float*)d_in[8];
    const float* W3 = (const float*)d_in[9];
    const float* b3 = (const float*)d_in[10];
    const float* Wp = (const float*)d_in[11];
    const float* bp = (const float*)d_in[12];
    float* out = (float*)d_out;

    char* p = (char*)d_ws;
    unsigned short* A0 = (unsigned short*)ws_take(p, (size_t)2048 * 256 * 2);
    unsigned short* B1 = (unsigned short*)ws_take(p, (size_t)256  * 512 * 2);
    float*          h1 = (float*)         ws_take(p, (size_t)2048 * 512 * 4);
    float*        mean1= (float*)         ws_take(p, 512 * 4);
    float*        rstd1= (float*)         ws_take(p, 512 * 4);
    unsigned short* A1 = (unsigned short*)ws_take(p, (size_t)2048 * 512 * 2);
    unsigned short* B2 = (unsigned short*)ws_take(p, (size_t)512  * 256 * 2);
    float*          h2 = (float*)         ws_take(p, (size_t)2048 * 256 * 4);
    float*        mean2= (float*)         ws_take(p, 256 * 4);
    float*        rstd2= (float*)         ws_take(p, 256 * 4);
    unsigned short* A2 = (unsigned short*)ws_take(p, (size_t)2048 * 256 * 2);
    unsigned short* B3 = (unsigned short*)ws_take(p, (size_t)256  * 112 * 2);
    float*        prm  = (float*)         ws_take(p, (size_t)2048 * 112 * 4);

    // ---- Layer 1: h1 = relu(x @ W1 + b1), M=2048 K=256 N=512 ----
    {   int total = 2048 * 256;                                   // A pack (x)
        pack_a_kernel<<<total / 256, 256, 0, stream>>>(x, A0, 256, 8, total); }
    {   int total = 32 * 8 * 512;                                 // B pack (W1)
        pack_b_kernel<<<(total + 255) / 256, 256, 0, stream>>>(W1, B1, 512, 8, total); }
    gemm_wmma_kernel<<<(128 * 32) / 8, 256, 0, stream>>>(A0, B1, b1, h1, 32, 8, 512, 512, 1);
    col_stats_kernel<<<512, 256, 0, stream>>>(h1, mean1, rstd1, 512);
    {   int total = 2048 * 512;                                   // BN + repack
        bn_pack_kernel<<<total / 256, 256, 0, stream>>>(h1, g1, be1, mean1, rstd1, A1, 512, 16, total); }

    // ---- Layer 2: h2 = relu(h1n @ W2 + b2), M=2048 K=512 N=256 ----
    {   int total = 16 * 16 * 512;
        pack_b_kernel<<<(total + 255) / 256, 256, 0, stream>>>(W2, B2, 256, 16, total); }
    gemm_wmma_kernel<<<(128 * 16) / 8, 256, 0, stream>>>(A1, B2, b2, h2, 16, 16, 256, 256, 1);
    col_stats_kernel<<<256, 256, 0, stream>>>(h2, mean2, rstd2, 256);
    {   int total = 2048 * 256;
        bn_pack_kernel<<<total / 256, 256, 0, stream>>>(h2, g2, be2, mean2, rstd2, A2, 256, 8, total); }

    // ---- Layer 3: params = h2n @ W3 + b3, M=2048 K=256 N=108 (pad 112) ----
    {   int total = 7 * 8 * 512;
        pack_b_kernel<<<(total + 255) / 256, 256, 0, stream>>>(W3, B3, 108, 8, total); }
    gemm_wmma_kernel<<<(128 * 7) / 8, 256, 0, stream>>>(A2, B3, b3, prm, 7, 8, 112, 108, 0);

    // ---- Quantum circuit + projection, one block per batch element ----
    quantum_kernel<<<2048, 256, 0, stream>>>(prm, Wp, bp, out);
}
